// CoattentiveLayer_28372553957398
// MI455X (gfx1250) — matrix operand order repaired and checked by reading, not compiled, not run on hardware
//
#include <hip/hip_runtime.h>
#include <hip/hip_bf16.h>
#include <math.h>

// ---------------- problem constants (fixed shapes) ----------------
#define B_   16
#define P_   1024
#define Q_   512
#define PD_  768
#define QD_  768
#define CD_  1536      // QD_ + PD_
#define HUGE_F 1e8f

// ---------------- WMMA vector types ----------------
typedef __attribute__((ext_vector_type(16))) __bf16          v16bf;
typedef __attribute__((ext_vector_type(2)))  __bf16          v2bf;
typedef __attribute__((ext_vector_type(8)))  float           v8f;
typedef __attribute__((ext_vector_type(16))) unsigned short  v16us;

union Frag { v16us u; v16bf b; };

// Native bf16 conversion -> v_cvt_pk_bf16_f32 (verified in round 3 asm).
__device__ __forceinline__ unsigned int pack2bf(float lo, float hi) {
    v2bf t;
    t.x = (__bf16)lo;
    t.y = (__bf16)hi;
    unsigned int r;
    __builtin_memcpy(&r, &t, 4);
    return r;
}

// =====================================================================
// Fully-static batched GEMM:  C[bz,m,n] = sum_k A(bz,m,k) * B(bz,k,n)
// All strides / K compile-time -> immediate-offset global loads, merged
// b64 loads on contiguous strides, single pointer bump per K-step.
// fp32 global -> bf16 LDS -> v_wmma_f32_16x16x32_bf16 -> fp32 out.
// Block tile 128x64x32, 256 threads = 8 waves, each wave 32x32 (4 WMMA).
// Double-buffered LDS, register-staged global tile, 1 barrier / K-step.
// REQUIRES: M % 128 == 0, N % 64 == 0, K % 32 == 0 (true for all calls).
// =====================================================================
#define BM  128
#define BN  64
#define BK  32
#define LDP 40     // padded LDS row in halves (80 B, 16B-aligned chunks)

template <int KT,
          long long SAB, int SAM, int SAK,
          long long SBB, int SBK, int SBN,
          long long SCB, int LDC, int COLOFF>
__global__ __launch_bounds__(256)
void gemm_bf16_wmma(const float* __restrict__ A, const float* __restrict__ Bm,
                    float* __restrict__ C)
{
    __shared__ unsigned short As[2][BM][LDP];   // [buf][m_local][k_local]
    __shared__ unsigned short Bs[2][BN][LDP];   // [buf][n_local][k_local]

    const int tid  = threadIdx.x;
    const int lane = tid & 31;
    const int wave = tid >> 5;
    const int wm   = wave & 3;        // 0..3 : 32-row slab of 128
    const int wn   = wave >> 2;       // 0..1 : 32-col slab of 64
    const int bz   = blockIdx.z;
    const int m0   = blockIdx.x * BM;
    const int n0   = blockIdx.y * BN;

    // staging thread mapping: k-pair kp..kp+1, rows ra + 16*i
    const int kp = (tid & 15) * 2;
    const int ra = tid >> 4;          // 0..15

    const float* pA = A  + (long long)bz * SAB + (long long)(m0 + ra) * SAM + (long long)kp * SAK;
    const float* pB = Bm + (long long)bz * SBB + (long long)(n0 + ra) * SBN + (long long)kp * SBK;

    v8f acc00 = {}, acc01 = {}, acc10 = {}, acc11 = {};

    const int hs   = (lane < 16) ? 0 : 1;   // half-wave select
    const int mrow = lane & 15;

    float av[16], bv[8];
    // prologue: tile 0 in flight (constant offsets -> immediate-form loads)
    #pragma unroll
    for (int i = 0; i < 8; ++i) {
        av[2 * i]     = pA[i * 16 * SAM];
        av[2 * i + 1] = pA[i * 16 * SAM + SAK];
    }
    #pragma unroll
    for (int i = 0; i < 4; ++i) {
        bv[2 * i]     = pB[i * 16 * SBN];
        bv[2 * i + 1] = pB[i * 16 * SBN + SBK];
    }

    int pp = 0;
    for (int k0 = 0; k0 < KT; k0 += BK) {
        // ---- convert + store staged tile to LDS[pp] (packed b32 stores) ----
        #pragma unroll
        for (int i = 0; i < 8; ++i)
            *(unsigned int*)&As[pp][ra + 16 * i][kp] = pack2bf(av[2 * i], av[2 * i + 1]);
        #pragma unroll
        for (int i = 0; i < 4; ++i)
            *(unsigned int*)&Bs[pp][ra + 16 * i][kp] = pack2bf(bv[2 * i], bv[2 * i + 1]);
        __syncthreads();

        // ---- issue next global tile before computing (overlap) ----
        pA += BK * SAK;
        pB += BK * SBK;
        if (k0 + BK < KT) {
            #pragma unroll
            for (int i = 0; i < 8; ++i) {
                av[2 * i]     = pA[i * 16 * SAM];
                av[2 * i + 1] = pA[i * 16 * SAM + SAK];
            }
            #pragma unroll
            for (int i = 0; i < 4; ++i) {
                bv[2 * i]     = pB[i * 16 * SBN];
                bv[2 * i + 1] = pB[i * 16 * SBN + SBK];
            }
        }

        // ---- fragments (documented CDNA5 16-bit layouts) ----
        Frag fa0, fa1, fb0, fb1;
        {
            const unsigned short* a0 = &As[pp][wm * 32 +      mrow][hs * 8];
            const unsigned short* a1 = &As[pp][wm * 32 + 16 + mrow][hs * 8];
            #pragma unroll
            for (int j = 0; j < 16; ++j) {
                int kk = (j < 8) ? j : (j + 8);
                fa0.u[j] = a0[kk];
                fa1.u[j] = a1[kk];
            }
            const unsigned short* b0 = &Bs[pp][wn * 32 +      mrow][hs * 8];
            const unsigned short* b1 = &Bs[pp][wn * 32 + 16 + mrow][hs * 8];
            #pragma unroll
            for (int j = 0; j < 16; ++j) {
                int kk = (j < 8) ? j : (j + 8);
                fb0.u[j] = b0[kk];
                fb1.u[j] = b1[kk];
            }
        }
        acc00 = __builtin_amdgcn_wmma_f32_16x16x32_bf16(false, fa0.b, false, fb0.b, (short)0, acc00, false, false);
        acc01 = __builtin_amdgcn_wmma_f32_16x16x32_bf16(false, fa0.b, false, fb1.b, (short)0, acc01, false, false);
        acc10 = __builtin_amdgcn_wmma_f32_16x16x32_bf16(false, fa1.b, false, fb0.b, (short)0, acc10, false, false);
        acc11 = __builtin_amdgcn_wmma_f32_16x16x32_bf16(false, fa1.b, false, fb1.b, (short)0, acc11, false, false);

        pp ^= 1;   // double buffer: next stores go to the other half
    }

    // ---- write C (f32), documented 16x16 f32 C/D layout ----
    float* Cb = C + (long long)bz * SCB + COLOFF;
    const int nb = n0 + wn * 32 + mrow;
    #pragma unroll
    for (int r = 0; r < 8; ++r) {
        const int ma = m0 + wm * 32 + r + hs * 8;
        Cb[(long long)ma * LDC + nb]             = acc00[r];
        Cb[(long long)ma * LDC + nb + 16]        = acc01[r];
        Cb[(long long)(ma + 16) * LDC + nb]      = acc10[r];
        Cb[(long long)(ma + 16) * LDC + nb + 16] = acc11[r];
    }
}

// =====================================================================
// Row softmax over contiguous q axis:
//   out[b,p,q] = softmax_q( q_mask[b,q] ? -HUGE : aff[b,p,q] )
// This IS attn_over_q^T in [B,P,Q] layout. One wave per row (Q=512).
// =====================================================================
__global__ __launch_bounds__(256)
void softmax_rows(const float* __restrict__ aff,
                  const unsigned char* __restrict__ qmask,
                  float* __restrict__ out)
{
    const int lane = threadIdx.x & 31;
    const int wave = threadIdx.x >> 5;
    const long long row = (long long)blockIdx.x * 8 + wave;   // b*P + p
    const int b = (int)(row >> 10);                           // P_ = 1024
    const float* r = aff + row * Q_;
    const unsigned char* qm = qmask + b * Q_;

    float v[16];
    float mx = -INFINITY;
    #pragma unroll
    for (int i = 0; i < 16; ++i) {
        int q = lane + i * 32;
        float x = qm[q] ? -HUGE_F : r[q];
        v[i] = x;
        mx = fmaxf(mx, x);
    }
    #pragma unroll
    for (int off = 16; off > 0; off >>= 1) mx = fmaxf(mx, __shfl_xor(mx, off, 32));

    float s = 0.0f;
    #pragma unroll
    for (int i = 0; i < 16; ++i) { v[i] = __expf(v[i] - mx); s += v[i]; }
    #pragma unroll
    for (int off = 16; off > 0; off >>= 1) s += __shfl_xor(s, off, 32);

    const float inv = 1.0f / s;
    float* o = out + row * Q_;
    #pragma unroll
    for (int i = 0; i < 16; ++i) o[lane + i * 32] = v[i] * inv;
}

// =====================================================================
// Column softmax over strided p axis, IN PLACE (-> attn_over_p):
//   aff[b,p,q] = softmax_p( p_mask[b,p] ? -HUGE : aff[b,p,q] )
// One thread per column q; online max/sum pass + write pass (L2 resident).
// =====================================================================
__global__ __launch_bounds__(256)
void softmax_cols_inplace(float* __restrict__ aff,
                          const unsigned char* __restrict__ pmask)
{
    const int b = blockIdx.y;
    const int q = blockIdx.x * 256 + threadIdx.x;
    float* base = aff + (long long)b * P_ * Q_ + q;
    const unsigned char* pm = pmask + b * P_;

    float m = -INFINITY, l = 0.0f;
    for (int p = 0; p < P_; ++p) {
        float x  = pm[p] ? -HUGE_F : base[(long long)p * Q_];
        float nm = fmaxf(m, x);
        l = l * __expf(m - nm) + __expf(x - nm);
        m = nm;
    }
    const float inv = 1.0f / l;
    for (int p = 0; p < P_; ++p) {
        float x = pm[p] ? -HUGE_F : base[(long long)p * Q_];
        base[(long long)p * Q_] = __expf(x - m) * inv;
    }
}

// =====================================================================
// Orchestration
// =====================================================================
extern "C" void kernel_launch(void* const* d_in, const int* in_sizes, int n_in,
                              void* d_out, int out_size, void* d_ws, size_t ws_size,
                              hipStream_t stream)
{
    (void)in_sizes; (void)n_in; (void)out_size; (void)ws_size;

    const float*         p   = (const float*)d_in[0];          // [16,1024,768]
    const float*         q   = (const float*)d_in[1];          // [16,512,768]
    const unsigned char* pmk = (const unsigned char*)d_in[2];  // [16,1024] bool
    const unsigned char* qmk = (const unsigned char*)d_in[3];  // [16,512]  bool
    const float*         W   = (const float*)d_in[4];          // [768,768]

    float* out1 = (float*)d_out;                               // p_to_q [16,1024,1536]
    float* out2 = out1 + (long long)B_ * P_ * CD_;             // q_to_p [16,512,768]

    float* wsf   = (float*)d_ws;
    float* qproj = wsf;                                        // [16,512,768]
    float* aff   = qproj + (long long)B_ * Q_ * PD_;           // [16,1024,512]
    float* smq   = aff   + (long long)B_ * P_ * Q_;            // attn_over_q^T [16,1024,512]

    dim3 blk(256);

    // 1) q_proj[m=b*Q+q, n=pd] = sum_k q[m,k] * W[n,k]     (M=8192,N=768,K=768)
    gemm_bf16_wmma<QD_, 0LL, QD_, 1, 0LL, 1, QD_, 0LL, PD_, 0>
        <<<dim3((B_*Q_)/BM, PD_/BN, 1), blk, 0, stream>>>(q, W, qproj);

    // 2) affinity[b,p,q] = sum_d p[b,p,d] * qproj[b,q,d]   (M=1024,N=512,K=768, bz=16)
    gemm_bf16_wmma<PD_, (long long)P_*PD_, PD_, 1,
                        (long long)Q_*PD_, 1, PD_,
                        (long long)P_*Q_, Q_, 0>
        <<<dim3(P_/BM, Q_/BN, B_), blk, 0, stream>>>(p, qproj, aff);

    // 3) attn_over_q^T = row softmax over q (with q_masks)
    softmax_rows<<<dim3((B_*P_)/8), blk, 0, stream>>>(aff, qmk, smq);

    // 4) attn_over_p = column softmax over p (with p_masks), in place
    softmax_cols_inplace<<<dim3(Q_/256, B_), blk, 0, stream>>>(aff, pmk);

    // 5) q_to_p[b,q,d] = sum_p attnP[b,p,q] * p[b,p,d]     (M=512,N=768,K=1024, bz=16)
    gemm_bf16_wmma<P_, (long long)P_*Q_, 1, Q_,
                       (long long)P_*PD_, PD_, 1,
                       (long long)Q_*PD_, PD_, 0>
        <<<dim3(Q_/BM, PD_/BN, B_), blk, 0, stream>>>(aff, p, out2);

    // 6a) p_to_q[:, 0:768]   = sum_q attnQ^T[b,p,q] * q[b,q,c]      (M=1024,N=768,K=512)
    gemm_bf16_wmma<Q_, (long long)P_*Q_, Q_, 1,
                       (long long)Q_*QD_, QD_, 1,
                       (long long)P_*CD_, CD_, 0>
        <<<dim3(P_/BM, QD_/BN, B_), blk, 0, stream>>>(smq, q, out1);

    // 6b) p_to_q[:, 768:1536] = sum_q attnQ^T[b,p,q] * q_to_p[b,q,c]
    gemm_bf16_wmma<Q_, (long long)P_*Q_, Q_, 1,
                       (long long)Q_*PD_, PD_, 1,
                       (long long)P_*CD_, CD_, PD_>
        <<<dim3(P_/BM, PD_/BN, B_), blk, 0, stream>>>(smq, out2, out1);
}
